// MyModel_86749749445008
// MI455X (gfx1250) — matrix operand order repaired and checked by reading, not compiled, not run on hardware
//
#include <hip/hip_runtime.h>
#include <hip/hip_bf16.h>
#include <math.h>

typedef __attribute__((ext_vector_type(16))) _Float16 v16h;
typedef __attribute__((ext_vector_type(8)))  _Float16 v8h;
typedef __attribute__((ext_vector_type(8)))  float    v8f;
typedef __attribute__((ext_vector_type(4)))  int      v4i;

#define BSZ   32
#define NNODE 1024
#define EDIM  64
#define UDIM  32
#define LITD  12
#define SEMD  16
#define TFWD  5
#define ROWP  (NNODE + 8)   // padded LDS row length (halves): 2064 B = 516 dwords -> conflict-free

static __device__ __forceinline__ float sigmoidf_(float x){ return 1.f/(1.f+__expf(-x)); }

// ---------------------------------------------------------------------------
// Prep: mlpW (2,64,64) f32 -> transposed f16  WT[l][j][i] = mlpW[l][i][j]
// ---------------------------------------------------------------------------
__global__ __launch_bounds__(256) void k_weights(const float* __restrict__ mlpW,
                                                 _Float16* __restrict__ wT) {
    for (int t = threadIdx.x; t < 2*64*64; t += 256) {
        int l = t >> 12, rem = t & 4095, j = rem >> 6, i = rem & 63;
        wT[l*4096 + j*64 + i] = (_Float16)mlpW[l*4096 + i*64 + j];
    }
}

// ---------------------------------------------------------------------------
// Literal path: mid = relu(lit@Win+bin); literal = relu(mid@Wout+bout)
// one 64-thread block per (b,n); writes bb[...,0:32]
// ---------------------------------------------------------------------------
__global__ __launch_bounds__(64) void k_literal(const float* __restrict__ lit,
                                                const float* __restrict__ Win,
                                                const float* __restrict__ bin,
                                                const float* __restrict__ Wout,
                                                const float* __restrict__ bout,
                                                float* __restrict__ bb) {
    int bn = blockIdx.x;                       // b*1024 + n
    __shared__ float xs[LITD];
    __shared__ float mid[EDIM];
    if (threadIdx.x < LITD) xs[threadIdx.x] = lit[(size_t)bn*LITD + threadIdx.x];
    __syncthreads();
    float s = bin[threadIdx.x];
    #pragma unroll
    for (int i = 0; i < LITD; ++i) s += xs[i] * Win[i*EDIM + threadIdx.x];
    mid[threadIdx.x] = fmaxf(s, 0.f);
    __syncthreads();
    if (threadIdx.x < UDIM) {
        float s2 = bout[threadIdx.x];
        #pragma unroll
        for (int i = 0; i < EDIM; ++i) s2 += mid[i] * Wout[i*UDIM + threadIdx.x];
        bb[(size_t)bn*EDIM + threadIdx.x] = fmaxf(s2, 0.f);
    }
}

// ---------------------------------------------------------------------------
// GRU input projection: xproj = sem @ gk + b_in     [B,N,96]
// ---------------------------------------------------------------------------
__global__ __launch_bounds__(256) void k_xproj(const float* __restrict__ sem,
                                               const float* __restrict__ gk,
                                               const float* __restrict__ gb,
                                               float* __restrict__ xproj) {
    int t = blockIdx.x*256 + threadIdx.x;
    if (t >= BSZ*NNODE*96) return;
    int c = t % 96, bn = t / 96;
    float s = gb[c];                           // b_in = gb[0][c]
    #pragma unroll
    for (int i = 0; i < SEMD; ++i) s += sem[(size_t)bn*SEMD + i] * gk[i*96 + c];
    xproj[t] = s;
}

// ---------------------------------------------------------------------------
// GRU recurrence (reset_after=True): one wave32 per batch row, h in registers,
// broadcast via __shfl. Writes semantic half bb[...,32:64].
// ---------------------------------------------------------------------------
__global__ __launch_bounds__(32) void k_gru(const float* __restrict__ xproj,
                                            const float* __restrict__ grk,
                                            const float* __restrict__ gb,
                                            float* __restrict__ bb) {
    int b = blockIdx.x, lane = threadIdx.x;
    float wz[UDIM], wr[UDIM], wh[UDIM];
    #pragma unroll
    for (int i = 0; i < UDIM; ++i) {
        wz[i] = grk[i*96 + lane];
        wr[i] = grk[i*96 + 32 + lane];
        wh[i] = grk[i*96 + 64 + lane];
    }
    float brz = gb[96 + lane], brr = gb[96 + 32 + lane], brh = gb[96 + 64 + lane];
    float h = 0.f;
    for (int t = 0; t < NNODE; ++t) {
        const float* xt = xproj + ((size_t)b*NNODE + t)*96;
        float rz = brz, rr = brr, rh = brh;
        #pragma unroll
        for (int i = 0; i < UDIM; ++i) {
            float hi = __shfl(h, i, 32);
            rz += hi*wz[i]; rr += hi*wr[i]; rh += hi*wh[i];
        }
        float z  = sigmoidf_(xt[lane] + rz);
        float r  = sigmoidf_(xt[32 + lane] + rr);
        float hh = tanhf(xt[64 + lane] + r*rh);
        h = z*h + (1.f - z)*hh;
        bb[((size_t)b*NNODE + t)*EDIM + UDIM + lane] = h;
    }
}

// ---------------------------------------------------------------------------
// e = relu(bb) in place
// ---------------------------------------------------------------------------
__global__ __launch_bounds__(256) void k_relu(float* __restrict__ e) {
    int t = blockIdx.x*256 + threadIdx.x;
    e[t] = fmaxf(e[t], 0.f);
}

// ---------------------------------------------------------------------------
// eT[b][j][n] = (f16) e[b][n][j]   -> makes WMMA B-fragments contiguous in K
// ---------------------------------------------------------------------------
__global__ __launch_bounds__(256) void k_transpose(const float* __restrict__ e,
                                                   _Float16* __restrict__ eT) {
    int t = blockIdx.x*256 + threadIdx.x;       // 32*64*1024 threads
    int n = t & 1023, j = (t >> 10) & 63, b = t >> 16;
    eT[((size_t)b*EDIM + j)*NNODE + n] = (_Float16)e[((size_t)b*NNODE + n)*EDIM + j];
}

// ---------------------------------------------------------------------------
// One graph-layer iteration, fully fused:
//   - async-stage eT[b] (64x1024 f16, 128 KB) into LDS once per workgroup
//     (global_load_async_to_lds_b128 + s_wait_asynccnt)
//   t = cfg @ e           (WMMA f16, K=1024, on-the-fly int->f16 for A,
//                          B fragments from LDS via ds_load_b128)
//   t = relu(t@W1 + b1) ; t = relu(t@W2 + b2)   (WMMA via LDS-staged tile)
//   e = tanh(e + t)
// grid (B, 8) x 256 threads; each wave owns a 16x64 output tile.
// ---------------------------------------------------------------------------
__global__ __launch_bounds__(256) void k_graph_iter(const int* __restrict__ cfg,
                                                    const _Float16* __restrict__ eT,
                                                    const float* __restrict__ e_in,
                                                    float* __restrict__ e_out,
                                                    const _Float16* __restrict__ W1T,
                                                    const _Float16* __restrict__ W2T,
                                                    const float* __restrict__ mlpb) {
    extern __shared__ __align__(16) _Float16 eTs[];          // [EDIM][ROWP] dynamic, 129 KB
    __shared__ __align__(16) _Float16 smem[8][16][EDIM];     // per-wave 16x64 f16 tile, 16 KB

    const int b    = blockIdx.x;
    const int slab = blockIdx.y;
    const int lane = threadIdx.x & 31;
    const int wave = threadIdx.x >> 5;
    const int m0   = slab*128 + wave*16;

    const int colBase = lane & 15;              // C-layout column within tile
    const int rowBase = (lane >> 4) * 8;        // C-layout row base (0 or 8)
    const int koff    = (lane >> 4) * 8;        // A-layout K interleave (0 or 8)
    const int kb      = (lane >> 4) * 16;       // unused (0 or 16); B loads use koff like A

    // ---- stage eT[b] into LDS via CDNA5 async copies ----
    {
        const _Float16* eTb = eT + (size_t)b*EDIM*NNODE;
        for (int c = threadIdx.x; c < EDIM*(NNODE/8); c += 256) {   // 8192 x 16B chunks
            int row = c >> 7, ch = c & 127;
            uint32_t ldsOff = (uint32_t)(uintptr_t)&eTs[row*ROWP + ch*8];
            uint64_t gaddr  = (uint64_t)(uintptr_t)(eTb + row*NNODE + ch*8);
            asm volatile("global_load_async_to_lds_b128 %0, %1, off"
                         :: "v"(ldsOff), "v"(gaddr) : "memory");
        }
        asm volatile("s_wait_asynccnt 0x0" ::: "memory");
    }
    __syncthreads();

    const int  mrow   = m0 + colBase;           // A-fragment row M = lane&15
    const int* cfgRow = cfg + ((size_t)b*NNODE + mrow)*NNODE;

    v8f acc[4] = {v8f{}, v8f{}, v8f{}, v8f{}};

    // ---- message passing: t = cfg_row_tile @ e ----
    for (int k0 = 0; k0 < NNODE; k0 += 32) {
        __builtin_prefetch(cfgRow + k0 + 512, 0, 0);
        // A fragment (16x32 f16): lanes 0-15 K=[k0..k0+7]+[k0+16..k0+23], lanes 16-31 shifted by 8
        const v4i* ap0 = (const v4i*)(cfgRow + k0 + koff);
        const v4i* ap1 = (const v4i*)(cfgRow + k0 + 16 + koff);
        v4i a0 = ap0[0], a1 = ap0[1], a2 = ap1[0], a3 = ap1[1];
        v16h af;
        #pragma unroll
        for (int i = 0; i < 4; ++i) {
            af[i]      = (_Float16)a0[i];
            af[4 + i]  = (_Float16)a1[i];
            af[8 + i]  = (_Float16)a2[i];
            af[12 + i] = (_Float16)a3[i];
        }
        #pragma unroll
        for (int nt = 0; nt < 4; ++nt) {
            // B fragment (32x16 f16) from LDS: lane N=lane&15,
            // lanes 0-15 K=[k0..k0+7]+[k0+16..k0+23], lanes 16-31 shifted by 8
            const _Float16* col = &eTs[(nt*16 + colBase)*ROWP + k0 + koff];
            v8h blo = *(const v8h*)(col);
            v8h bhi = *(const v8h*)(col + 16);
            v16h bf;
            #pragma unroll
            for (int i = 0; i < 8; ++i) { bf[i] = blo[i]; bf[8 + i] = bhi[i]; }
            acc[nt] = __builtin_amdgcn_wmma_f32_16x16x32_f16(false, af, false, bf,
                                                             (short)0, acc[nt], false, false);
        }
    }

    // ---- MLP layer 1 & 2 via WMMA, staging f16 tile in LDS ----
    const _Float16* WT[2] = {W1T, W2T};
    #pragma unroll
    for (int layer = 0; layer < 2; ++layer) {
        #pragma unroll
        for (int nt = 0; nt < 4; ++nt)
            #pragma unroll
            for (int r = 0; r < 8; ++r)
                smem[wave][rowBase + r][nt*16 + colBase] = (_Float16)acc[nt][r];
        __syncthreads();

        v8f acc2[4] = {v8f{}, v8f{}, v8f{}, v8f{}};
        const _Float16* lrow = &smem[wave][colBase][0];   // A row M = lane&15
        #pragma unroll
        for (int k0 = 0; k0 < EDIM; k0 += 32) {
            v8h alo = *(const v8h*)(lrow + k0 + koff);
            v8h ahi = *(const v8h*)(lrow + k0 + 16 + koff);
            v16h af;
            #pragma unroll
            for (int i = 0; i < 8; ++i) { af[i] = alo[i]; af[8 + i] = ahi[i]; }
            #pragma unroll
            for (int nt = 0; nt < 4; ++nt) {
                const _Float16* wrow = WT[layer] + (size_t)(nt*16 + colBase)*EDIM + k0 + koff;
                v8h blo = *(const v8h*)(wrow);
                v8h bhi = *(const v8h*)(wrow + 16);
                v16h bf;
                #pragma unroll
                for (int i = 0; i < 8; ++i) { bf[i] = blo[i]; bf[8 + i] = bhi[i]; }
                acc2[nt] = __builtin_amdgcn_wmma_f32_16x16x32_f16(false, af, false, bf,
                                                                  (short)0, acc2[nt], false, false);
            }
        }
        // bias + relu
        #pragma unroll
        for (int nt = 0; nt < 4; ++nt) {
            float bj = mlpb[layer*EDIM + nt*16 + colBase];
            #pragma unroll
            for (int r = 0; r < 8; ++r) acc2[nt][r] = fmaxf(acc2[nt][r] + bj, 0.f);
        }
        #pragma unroll
        for (int nt = 0; nt < 4; ++nt) acc[nt] = acc2[nt];
        __syncthreads();
    }

    // ---- e = tanh(e + t) ----
    #pragma unroll
    for (int nt = 0; nt < 4; ++nt)
        #pragma unroll
        for (int r = 0; r < 8; ++r) {
            int m = m0 + rowBase + r;
            int n = nt*16 + colBase;
            size_t idx = ((size_t)b*NNODE + m)*EDIM + n;
            e_out[idx] = tanhf(e_in[idx] + acc[nt][r]);
        }
}

// ---------------------------------------------------------------------------
// mid = sum_n e[b,n,:];  out = mid @ wOut + bOut      (one 64-thread block / b)
// ---------------------------------------------------------------------------
__global__ __launch_bounds__(64) void k_reduce(const float* __restrict__ e,
                                               const float* __restrict__ wOut,
                                               const float* __restrict__ bOut,
                                               float* __restrict__ out) {
    int b = blockIdx.x, tid = threadIdx.x;
    __shared__ float midv[EDIM];
    float s = 0.f;
    for (int n = 0; n < NNODE; ++n) s += e[((size_t)b*NNODE + n)*EDIM + tid];
    midv[tid] = s;
    __syncthreads();
    float o = bOut[tid];
    #pragma unroll
    for (int k = 0; k < EDIM; ++k) o += midv[k] * wOut[k*EDIM + tid];
    out[(size_t)b*EDIM + tid] = o;
}

// ---------------------------------------------------------------------------
// label[b] = (1 + cos(e1,e2)) / 2
// ---------------------------------------------------------------------------
__global__ __launch_bounds__(32) void k_cosine(const float* __restrict__ o1,
                                               const float* __restrict__ o2,
                                               float* __restrict__ label) {
    int b = threadIdx.x;
    float dot = 0.f, na = 0.f, nb = 0.f;
    #pragma unroll
    for (int k = 0; k < EDIM; ++k) {
        float a = o1[(size_t)b*EDIM + k], c = o2[(size_t)b*EDIM + k];
        dot += a*c; na += a*a; nb += c*c;
    }
    float cs = dot / (sqrtf(fmaxf(na, 1e-12f)) * sqrtf(fmaxf(nb, 1e-12f)));
    label[b] = (1.f + cs) * 0.5f;
}

// ---------------------------------------------------------------------------
extern "C" void kernel_launch(void* const* d_in, const int* in_sizes, int n_in,
                              void* d_out, int out_size, void* d_ws, size_t ws_size,
                              hipStream_t stream) {
    const int*   cfg[2]  = {(const int*)d_in[0],  (const int*)d_in[4]};
    const float* lit[2]  = {(const float*)d_in[2],  (const float*)d_in[6]};
    const float* sem[2]  = {(const float*)d_in[3],  (const float*)d_in[7]};
    const float* Win[2]  = {(const float*)d_in[8],  (const float*)d_in[15]};
    const float* bin[2]  = {(const float*)d_in[9],  (const float*)d_in[16]};
    const float* Wout[2] = {(const float*)d_in[10], (const float*)d_in[17]};
    const float* bout[2] = {(const float*)d_in[11], (const float*)d_in[18]};
    const float* gk[2]   = {(const float*)d_in[12], (const float*)d_in[19]};
    const float* grk[2]  = {(const float*)d_in[13], (const float*)d_in[20]};
    const float* gb[2]   = {(const float*)d_in[14], (const float*)d_in[21]};
    const float* mlpW = (const float*)d_in[22];
    const float* mlpb = (const float*)d_in[23];
    const float* wOut = (const float*)d_in[24];
    const float* bOut = (const float*)d_in[25];

    char* ws = (char*)d_ws;
    float*    ebuf  = (float*)ws;                              //  8 MB  [B,N,64] f32
    _Float16* eT    = (_Float16*)(ws + (8u  << 20));           //  4 MB  [B,64,N] f16
    float*    xproj = (float*)   (ws + (12u << 20));           // 12 MB  [B,N,96] f32
    _Float16* wT    = (_Float16*)(ws + (24u << 20));           // 16 KB  W1T|W2T f16
    float*    ov0   = (float*)   (ws + (24u << 20) + (1u << 16));
    float*    ov1   = ov0 + BSZ*EDIM;
    float*    outv[2] = {ov0, ov1};

    const size_t dynLds = (size_t)EDIM * ROWP * sizeof(_Float16);   // 132,096 B

    k_weights<<<1, 256, 0, stream>>>(mlpW, wT);

    for (int g = 0; g < 2; ++g) {
        k_literal<<<BSZ*NNODE, 64, 0, stream>>>(lit[g], Win[g], bin[g], Wout[g], bout[g], ebuf);
        k_xproj<<<(BSZ*NNODE*96 + 255)/256, 256, 0, stream>>>(sem[g], gk[g], gb[g], xproj);
        k_gru<<<BSZ, 32, 0, stream>>>(xproj, grk[g], gb[g], ebuf);
        k_relu<<<(BSZ*NNODE*EDIM)/256, 256, 0, stream>>>(ebuf);
        for (int it = 0; it < TFWD; ++it) {
            k_transpose<<<(BSZ*NNODE*EDIM)/256, 256, 0, stream>>>(ebuf, eT);
            k_graph_iter<<<dim3(BSZ, 8), 256, dynLds, stream>>>(cfg[g], eT, ebuf, ebuf,
                                                                wT, wT + 4096, mlpb);
        }
        k_reduce<<<BSZ, 64, 0, stream>>>(ebuf, wOut, bOut, outv[g]);
    }
    k_cosine<<<1, 32, 0, stream>>>(outv[0], outv[1], (float*)d_out);
}
